// ModConv2D_5179730559359
// MI455X (gfx1250) — matrix operand order repaired and checked by reading, not compiled, not run on hardware
//
#include <hip/hip_runtime.h>
#include <hip/hip_bf16.h>

typedef __attribute__((ext_vector_type(16))) __bf16 v16bf;
typedef __attribute__((ext_vector_type(8)))  __bf16 v8bf;
typedef __attribute__((ext_vector_type(8)))  float  v8f;
typedef __attribute__((ext_vector_type(4)))  unsigned int v4u;
typedef __attribute__((ext_vector_type(8)))  int    v8i;
typedef __attribute__((ext_vector_type(4)))  int    v4i;

#define IN_DIM   512
#define OUT_DIM  512
#define HW       64
#define NSPAT    (HW * HW)          // 4096 spatial positions per image
#define KTOT     (IN_DIM * 9)       // 4608 = GEMM K
#define M_TILE   128
#define N_TILE   128
#define K_STEP   32
#define LDS_STRIDE 40               // bf16 per n-row: 32 used + 8 pad = 80B (bank-friendly)

// TDM builtin exists on both toolchains but with different arity:
//   ROCm 7.2 (clang-22): 5 args (g0, g1, g2, g3, cpol)
//   amdgpu-toolchain (clang-23): 6 args (g0, g1, g2, g3, g4, cpol)
#if defined(__has_builtin)
#  if __has_builtin(__builtin_amdgcn_tensor_load_to_lds)
#    define HAVE_TDM 1
#  endif
#endif
#ifndef HAVE_TDM
#  define HAVE_TDM 0
#endif

// ---------------------------------------------------------------------------
// Kernel 0: convert activations fp32 -> bf16 (one time; layout unchanged NCHW)
// ---------------------------------------------------------------------------
__global__ void cvt_x_bf16(const float* __restrict__ x, __bf16* __restrict__ xbf, int n) {
    int i = blockIdx.x * blockDim.x + threadIdx.x;
    if (i < n) xbf[i] = (__bf16)x[i];
}

// ---------------------------------------------------------------------------
// Kernel 1: modulate + demodulate weights per (batch, out-channel), emit bf16
// Wb[b][o][k], k = ci*9 + ky*3 + kx (contiguous, matches OIHW inner order)
// ---------------------------------------------------------------------------
__global__ void __launch_bounds__(256) modulate_w(const float* __restrict__ weight,
                                                  const float* __restrict__ style,
                                                  __bf16* __restrict__ wb) {
    __shared__ float wsh[KTOT];
    __shared__ float red[256];

    const int o = blockIdx.x;
    const int b = blockIdx.y;
    const int t = threadIdx.x;
    const float c = 0.014731391274719738f;          // 1/sqrt(512*3*3)

    const float* wr = weight + (size_t)o * KTOT;
    const float* st = style  + (size_t)b * IN_DIM;

    float ss = 0.0f;
    for (int k = t; k < KTOT; k += 256) {
        int ci = k / 9;
        float v = wr[k] * c * (st[ci] + 1.0f);
        wsh[k] = v;
        ss += v * v;
    }
    red[t] = ss;
    __syncthreads();
    for (int off = 128; off > 0; off >>= 1) {
        if (t < off) red[t] += red[t + off];
        __syncthreads();
    }
    const float sinv = __frsqrt_rn(red[0] + 1e-8f);

    __bf16* dst = wb + ((size_t)b * OUT_DIM + o) * KTOT;
    for (int k = t; k < KTOT; k += 256)
        dst[k] = (__bf16)(wsh[k] * sinv);
}

// ---------------------------------------------------------------------------
// Kernel 2: implicit-GEMM 3x3 SAME conv via v_wmma_f32_16x16x32_bf16.
// Per batch: C[512, 4096] = Wb[512, 4608] x im2col(x)[4608, 4096].
// Block = 8 waves; wave w computes rows [tile_m+16w, +16) x cols [tile_n, +128).
// A (weight) tile is DMA'd into LDS by the Tensor Data Mover each K-step.
// ---------------------------------------------------------------------------
__global__ void __launch_bounds__(256) modconv_wmma(const __bf16* __restrict__ xbf,
                                                    const __bf16* __restrict__ wb,
                                                    const float* __restrict__ bias,
                                                    float* __restrict__ out) {
    __shared__ __bf16 btile[N_TILE * LDS_STRIDE];   // im2col [n][k], 10 KB
    __shared__ __bf16 atile[M_TILE * K_STEP];       // weight tile [m][k], 8 KB (64B rows)

    const int b      = blockIdx.z;
    const int tile_m = blockIdx.y * M_TILE;
    const int tile_n = blockIdx.x * N_TILE;
    const int t      = threadIdx.x;
    const int wave   = t >> 5;
    const int lane   = t & 31;
    const int lhalf  = lane >> 4;                   // 0 | 1
    const int l16    = lane & 15;

    const int m_base = tile_m + wave * 16;

    v8f acc[8];
#pragma unroll
    for (int s = 0; s < 8; ++s) acc[s] = (v8f)0.0f;

    // --- B-tile build assignment: thread -> (k-line, 16-wide n run) ---
    const int kl  = t >> 3;                         // 0..31
    const int nb  = (t & 7) << 4;                   // 0,16,...,112
    const int ng0 = tile_n + nb;
    const int yb  = ng0 >> 6;                       // output row (runs of 16 never cross rows)
    const int xb  = ng0 & (HW - 1);                 // output col base

    const __bf16* xbatch = xbf + (size_t)b * IN_DIM * NSPAT;

#if HAVE_TDM
    // Loop-invariant D# group 1: data_size=2B, tensor_dim0=KTOT, tensor_dim1=4096,
    // tile = 32(k) x 128(m), tensor_dim0_stride = KTOT. (Fields per CDNA5 ISA 8.4)
    const v8i tdm_g1 = {
        0x00010000,                                  // workgroup_mask=0, data_size=1 (2B)
        (int)((KTOT & 0xFFFF) << 16),                // tensor_dim0[15:0] << 16
        (int)(((KTOT >> 16) & 0xFFFF) | (4096 << 16)), // dim0[31:16] | tensor_dim1[15:0]<<16
        (int)(K_STEP << 16),                         // dim1[31:16]=0 | tile_dim0<<16
        (int)M_TILE,                                 // tile_dim1 | tile_dim2<<16
        (int)KTOT,                                   // tensor_dim0_stride[31:0]
        0, 0                                         // stride hi / dim1_stride (unused, 2D)
    };
    const v4i tdm_g2 = { 0, 0, 0, 0 };
    const v4i tdm_g3 = { 0, 0, 0, 0 };
    const unsigned int lds_atile = (unsigned int)(uintptr_t)(void*)atile; // LDS byte offset
#else
    // Fallback: per documented 16-bit A layout, lanes<16 take K {0..7,16..23},
    // lanes>=16 take K {8..15,24..31} -> two 16B loads at +0/+16 (half shifted by 8).
    const __bf16* arow = wb + ((size_t)b * OUT_DIM + m_base + l16) * KTOT + (lhalf << 3);
#endif

    for (int k0 = 0; k0 < KTOT; k0 += K_STEP) {
        // ---- stage im2col B tile (32 K x 128 N) into LDS ----
        {
            int k  = k0 + kl;
            int ci = k / 9;
            int kk = k - ci * 9;
            int ky = kk / 3;
            int kx = kk - ky * 3;
            int ys = yb + ky - 1;
            bool yok = (ys >= 0) & (ys < HW);
            const __bf16* xc = xbatch + (size_t)ci * NSPAT + ys * HW;
#pragma unroll
            for (int j = 0; j < 16; ++j) {
                int xs = xb + j + kx - 1;
                __bf16 v = (yok & (xs >= 0) & (xs < HW)) ? xc[xs] : (__bf16)0.0f;
                btile[(nb + j) * LDS_STRIDE + kl] = v;
            }
        }

#if HAVE_TDM
        // ---- TDM: DMA the 128x32 bf16 weight tile into LDS (wave 0 issues) ----
        if (wave == 0) {
            unsigned long long ga = (unsigned long long)(uintptr_t)
                (wb + ((size_t)b * OUT_DIM + tile_m) * KTOT + k0);
            v4u g0 = { 1u,                              // count=1, no gather
                       lds_atile,                       // D#.lds_addr
                       (unsigned int)ga,                // global_addr[31:0]
                       (unsigned int)(ga >> 32) | (2u << 30) }; // addr[56:32] | type=2
#if __clang_major__ >= 23
            const v8i tdm_g4 = { 0, 0, 0, 0, 0, 0, 0, 0 };
            __builtin_amdgcn_tensor_load_to_lds(g0, tdm_g1, tdm_g2, tdm_g3, tdm_g4, 0);
#else
            __builtin_amdgcn_tensor_load_to_lds(g0, tdm_g1, tdm_g2, tdm_g3, 0);
#endif
            __builtin_amdgcn_s_wait_tensorcnt(0);
        }
#endif
        __syncthreads();

        // ---- A fragment ----
#if HAVE_TDM
        const __bf16* ap = &atile[(wave * 16 + l16) * K_STEP + (lhalf << 3)];
        v8bf a0 = *(const v8bf*)ap;
        v8bf a1 = *(const v8bf*)(ap + 16);
#else
        v8bf a0 = *(const v8bf*)(arow + k0);
        v8bf a1 = *(const v8bf*)(arow + k0 + 16);
        __builtin_prefetch(arow + k0 + K_STEP, 0, 0);   // global_prefetch_b8 next K-step
#endif
        v16bf afrag = __builtin_shufflevector(a0, a1,
            0, 1, 2, 3, 4, 5, 6, 7, 8, 9, 10, 11, 12, 13, 14, 15);

        // ---- 8 WMMAs, A reused across the 128-wide N strip ----
        const int koff = lhalf << 3;                    // B lanes>=16 hold K {8..15,24..31}
#pragma unroll
        for (int s = 0; s < 8; ++s) {
            const __bf16* bp = &btile[(s * 16 + l16) * LDS_STRIDE + koff];
            v8bf b0 = *(const v8bf*)bp;
            v8bf b1 = *(const v8bf*)(bp + 16);
            v16bf bfrag = __builtin_shufflevector(b0, b1,
                0, 1, 2, 3, 4, 5, 6, 7, 8, 9, 10, 11, 12, 13, 14, 15);
            acc[s] = __builtin_amdgcn_wmma_f32_16x16x32_bf16(
                /*neg_a=*/false, afrag, /*neg_b=*/false, bfrag,
                /*c_mod=*/(short)0, acc[s], /*reuse_a=*/false, /*reuse_b=*/false);
        }
        __syncthreads();
    }

    // ---- epilogue: C layout VGPR r -> M = r + 8*lhalf, N = l16 ----
    const int mrow = m_base + (lhalf << 3);
#pragma unroll
    for (int s = 0; s < 8; ++s) {
        int n = tile_n + s * 16 + l16;
        float* op = out + ((size_t)b * OUT_DIM + mrow) * NSPAT + n;
#pragma unroll
        for (int r = 0; r < 8; ++r)
            op[(size_t)r * NSPAT] = acc[s][r] + bias[mrow + r];
    }
}

// ---------------------------------------------------------------------------
// Launcher. d_ws layout: [0, 32MB) x as bf16; [32MB, 68MB) modulated W as bf16.
// ---------------------------------------------------------------------------
extern "C" void kernel_launch(void* const* d_in, const int* in_sizes, int n_in,
                              void* d_out, int out_size, void* d_ws, size_t ws_size,
                              hipStream_t stream) {
    const float* x      = (const float*)d_in[0];   // [8,512,64,64]
    const float* style  = (const float*)d_in[1];   // [8,512]
    const float* weight = (const float*)d_in[2];   // [512,512,3,3]
    const float* bias   = (const float*)d_in[3];   // [512]
    float* out = (float*)d_out;                    // [8,512,64,64]

    const int nx = in_sizes[0];                            // 16,777,216
    __bf16* xbf = (__bf16*)d_ws;
    __bf16* wbm = (__bf16*)((char*)d_ws + (size_t)nx * sizeof(__bf16));

    cvt_x_bf16<<<(nx + 255) / 256, 256, 0, stream>>>(x, xbf, nx);

    modulate_w<<<dim3(OUT_DIM, 8), 256, 0, stream>>>(weight, style, wbm);

    modconv_wmma<<<dim3(NSPAT / N_TILE, OUT_DIM / M_TILE, 8), 256, 0, stream>>>(
        xbf, wbm, bias, out);
}
